// ModernBertAttention_43147241456091
// MI455X (gfx1250) — matrix operand order repaired and checked by reading, not compiled, not run on hardware
//
#include <hip/hip_runtime.h>
#include <hip/hip_bf16.h>

// ---------------------------------------------------------------------------
// ModernBERT sliding-window attention for gfx1250 (MI455X), bf16 WMMA path.
//   B=2, S=2048, Dm=768, H=12, Dh=64, window=+-64
// GEMMs stage the shared B tile in LDS via async global->LDS (ASYNCcnt).
// ---------------------------------------------------------------------------

typedef __attribute__((ext_vector_type(8)))  float  v8f;
typedef __attribute__((ext_vector_type(16))) __bf16 v16bf;
typedef __attribute__((ext_vector_type(8)))  __bf16 v8bf;

union FragBF { v8bf h[2]; v16bf v; };

static __device__ __forceinline__ unsigned short f2bf(float f) {
    __bf16 h = (__bf16)f;
    return __builtin_bit_cast(unsigned short, h);
}

#define WMMA_BF16(A, B, C) \
    __builtin_amdgcn_wmma_f32_16x16x32_bf16(false, (A), false, (B), (short)0, (C), false, false)

// Async global->LDS 16B copy (per lane). ldsOff = LDS byte offset (addrspace-3
// pointers on AMDGPU are offsets from the wave's LDS base). Tracked by ASYNCcnt.
static __device__ __forceinline__ void async_g2l_b128(unsigned ldsOff, const unsigned short* g) {
    asm volatile("global_load_async_to_lds_b128 %0, %1, off"
                 :: "v"(ldsOff), "v"(g) : "memory");
}
static __device__ __forceinline__ void wait_async_2() { asm volatile("s_wait_asynccnt 0x2" ::: "memory"); }
static __device__ __forceinline__ void wait_async_0() { asm volatile("s_wait_asynccnt 0x0" ::: "memory"); }

static constexpr int Bc = 2, Sc = 2048, DMc = 768, Hc = 12;

// ---------------------------------------------------------------------------
// fp32 -> bf16 copy
// ---------------------------------------------------------------------------
__global__ void k_cvt(const float* __restrict__ in, unsigned short* __restrict__ out, int n) {
    int i = blockIdx.x * blockDim.x + threadIdx.x;
    if (i < n) out[i] = f2bf(in[i]);
}

// fp32 [rows][cols] -> bf16 [cols][rows] (transpose so B-operand rows are contiguous K)
__global__ void k_transpose_cvt(const float* __restrict__ in, unsigned short* __restrict__ out,
                                int rows, int cols) {
    int i = blockIdx.x * blockDim.x + threadIdx.x;
    if (i < rows * cols) {
        int r = i / cols, c = i % cols;
        out[(size_t)c * rows + r] = f2bf(in[i]);
    }
}

// ---------------------------------------------------------------------------
// GEMM: C[M][N] (f32) = A[M][K] (bf16) * Bt[N][K] (bf16, pre-transposed)
// Block = 128 threads (4 waves); each wave computes a 16x64 tile.
// The 64x32 B tile (shared by all 4 waves) is staged in LDS with async
// global->LDS copies, double buffered; A fragments come straight from global.
// All four B fragments are loaded into distinct registers before the WMMAs so
// the scheduler can overlap ds_load latency with matrix math (partial dscnt).
// M, N multiples of 64; K multiple of 32.
// ---------------------------------------------------------------------------
__global__ __launch_bounds__(128) void k_gemm(const unsigned short* __restrict__ A,
                                              const unsigned short* __restrict__ Bt,
                                              float* __restrict__ C,
                                              int M, int N, int K) {
    __shared__ __align__(16) unsigned short Btile[2][64 * 32];

    const int t    = threadIdx.x;
    const int lane = t & 31;
    const int wave = t >> 5;
    const int lo16 = lane & 15;
    const int hi   = lane >> 4;
    const int kb   = hi * 8;    // A fragment per-lane K base (0 or 8)
    const int ko   = hi * 16;   // unused; B fragments also use kb (0 or 8)

    const int row0 = blockIdx.y * 64 + wave * 16;
    const int col0 = blockIdx.x * 64;

    // Per-thread async-copy assignment: thread t stages row (t>>1) of the B
    // tile, half-column (t&1)*16 .. +15  (16 bf16 = 32B = 2 x b128).
    const int brow  = t >> 1;
    const int bhalf = (t & 1) * 16;
    const unsigned short* bsrc = Bt + (size_t)(col0 + brow) * K + bhalf;
    const unsigned ldsBase0 = (unsigned)(size_t)&Btile[0][brow * 32 + bhalf];
    const unsigned ldsBase1 = (unsigned)(size_t)&Btile[1][brow * 32 + bhalf];

    v8f acc[4] = {v8f{}, v8f{}, v8f{}, v8f{}};
    const unsigned short* arow = A + (size_t)(row0 + lo16) * K;

    const int nsteps = K >> 5;   // K/32

    // prologue: stage buffer 0 for k=0
    async_g2l_b128(ldsBase0,      bsrc);
    async_g2l_b128(ldsBase0 + 16, bsrc + 8);

    for (int i = 0; i < nsteps; ++i) {
        const int k = i << 5;
        if (i + 1 < nsteps) {
            // stage next k-step into the other buffer while we compute
            const unsigned nbase = ((i + 1) & 1) ? ldsBase1 : ldsBase0;
            const unsigned short* ns = bsrc + (k + 32);
            async_g2l_b128(nbase,      ns);
            async_g2l_b128(nbase + 16, ns + 8);
            wait_async_2();          // current buffer's 2 copies have landed
        } else {
            wait_async_0();
        }
        __syncthreads();             // all waves' copies visible

        __builtin_prefetch(arow + k + 128, 0, 0);   // global_prefetch_b8 (A stream)
        FragBF fa;
        fa.h[0] = *(const v8bf*)(arow + k + kb);
        fa.h[1] = *(const v8bf*)(arow + k + kb + 16);

        const unsigned short* btile = Btile[i & 1];
        FragBF fb[4];
#pragma unroll
        for (int nt = 0; nt < 4; ++nt) {
            const unsigned short* bp = btile + (nt * 16 + lo16) * 32 + kb;
            fb[nt].h[0] = *(const v8bf*)(bp);
            fb[nt].h[1] = *(const v8bf*)(bp + 16);
        }
#pragma unroll
        for (int nt = 0; nt < 4; ++nt)
            acc[nt] = WMMA_BF16(fa.v, fb[nt].v, acc[nt]);

        __syncthreads();             // done reading before buffer is re-staged
    }

#pragma unroll
    for (int nt = 0; nt < 4; ++nt) {
#pragma unroll
        for (int r = 0; r < 8; ++r) {
            const int row = row0 + hi * 8 + r;
            const int col = col0 + nt * 16 + lo16;
            C[(size_t)row * N + col] = acc[nt][r];
        }
    }
}

// ---------------------------------------------------------------------------
// RoPE + scatter: qkv f32 [B*S][2304] -> Q,K bf16 [b,h,s,64] (roped),
//                 V bf16 transposed [b,h,d][s] for the PV WMMA B-operand.
// One thread per (b,s,h,j) with j in [0,32): handles the rotate pair (j, j+32).
// ---------------------------------------------------------------------------
__global__ void k_rope(const float* __restrict__ qkv,
                       unsigned short* __restrict__ Q,
                       unsigned short* __restrict__ Kk,
                       unsigned short* __restrict__ Vt) {
    int i = blockIdx.x * blockDim.x + threadIdx.x;
    const int j = i & 31;
    const int h = (i >> 5) % Hc;
    const int s = (i / (32 * Hc)) % Sc;
    const int b = i / (32 * Hc * Sc);
    if (b >= Bc) return;

    const float* row = qkv + (size_t)(b * Sc + s) * (3 * DMc);
    // inv_freq = 10000^(-j/32) = exp(-j * ln(1e4)/32)
    const float inv_freq = __expf(-(float)j * 0.28782313662425575f);
    float sn, cs;
    __sincosf((float)s * inv_freq, &sn, &cs);

    const int qb = h * 64, kb = DMc + h * 64, vb = 2 * DMc + h * 64;
    const float q1 = row[qb + j], q2 = row[qb + 32 + j];
    const float k1 = row[kb + j], k2 = row[kb + 32 + j];
    const float v1 = row[vb + j], v2 = row[vb + 32 + j];

    unsigned short* qo = Q  + ((size_t)((b * Hc + h) * Sc) + s) * 64;
    unsigned short* ko = Kk + ((size_t)((b * Hc + h) * Sc) + s) * 64;
    qo[j]      = f2bf(q1 * cs - q2 * sn);
    qo[j + 32] = f2bf(q2 * cs + q1 * sn);
    ko[j]      = f2bf(k1 * cs - k2 * sn);
    ko[j + 32] = f2bf(k2 * cs + k1 * sn);

    unsigned short* vo = Vt + (size_t)((b * Hc + h) * 64) * Sc;
    vo[(size_t)j * Sc + s]        = f2bf(v1);
    vo[(size_t)(j + 32) * Sc + s] = f2bf(v2);
}

// ---------------------------------------------------------------------------
// Sliding-window attention, one wave (32 threads) per 16-query tile per (b,h).
// Key window: 9 16-aligned tiles covering [q0-64, q0+80); distance mask kills
// everything outside +-64 (exp -> 0), so padding tiles contribute nothing.
// scores: 18 WMMAs; softmax in registers with shfl_xor over 16-lane groups;
// P staged bf16 in LDS (16x160, zero padded); PV: 20 WMMAs.
// Output O bf16 [B*S][768].
// ---------------------------------------------------------------------------
__global__ __launch_bounds__(32) void k_attn(const unsigned short* __restrict__ Qg,
                                             const unsigned short* __restrict__ Kg,
                                             const unsigned short* __restrict__ Vtg,
                                             unsigned short* __restrict__ Og) {
    const int lane = threadIdx.x & 31;
    const int id = blockIdx.x;
    const int qt = id & 127;                  // S/16 = 128 query tiles
    const int h  = (id >> 7) % Hc;
    const int b  = id / (128 * Hc);
    const int q0 = qt * 16;
    const int s0 = (q0 >= 64) ? (q0 - 64) : 0;   // 16-aligned window start

    __shared__ __align__(16) unsigned short P[16 * 160];
    for (int i = lane; i < 16 * 160; i += 32) P[i] = 0;
    __syncthreads();

    const int lo16 = lane & 15;
    const int hi   = lane >> 4;
    const int kb   = hi * 8;
    const int ko   = hi * 16;

    // Q fragments for d=[0,32) and d=[32,64)
    const unsigned short* qrow = Qg + ((size_t)((b * Hc + h) * Sc) + q0 + lo16) * 64;
    FragBF aq0, aq1;
    aq0.h[0] = *(const v8bf*)(qrow + kb);
    aq0.h[1] = *(const v8bf*)(qrow + kb + 16);
    aq1.h[0] = *(const v8bf*)(qrow + 32 + kb);
    aq1.h[1] = *(const v8bf*)(qrow + 32 + kb + 16);

    const unsigned short* kmat = Kg + (size_t)((b * Hc + h) * Sc) * 64;

    float sc[9][8];
#pragma unroll
    for (int kt = 0; kt < 9; ++kt) {
        const int key = s0 + kt * 16 + lo16;       // this lane's key column
        const unsigned short* kp = kmat + (size_t)key * 64;
        FragBF b0, b1;
        b0.h[0] = *(const v8bf*)(kp + kb);
        b0.h[1] = *(const v8bf*)(kp + kb + 16);
        b1.h[0] = *(const v8bf*)(kp + 32 + kb);
        b1.h[1] = *(const v8bf*)(kp + 32 + kb + 16);
        v8f c = {};
        c = WMMA_BF16(aq0.v, b0.v, c);
        c = WMMA_BF16(aq1.v, b1.v, c);
#pragma unroll
        for (int r = 0; r < 8; ++r) {
            const int qpos = q0 + hi * 8 + r;
            const int dist = key - qpos;
            sc[kt][r] = (dist <= 64 && dist >= -64) ? c[r] * 0.125f : -1e30f;
        }
    }

    // row-wise softmax (rows live across 16-lane groups)
#pragma unroll
    for (int r = 0; r < 8; ++r) {
        float m = -1e30f;
#pragma unroll
        for (int kt = 0; kt < 9; ++kt) m = fmaxf(m, sc[kt][r]);
        m = fmaxf(m, __shfl_xor(m, 1, 16));
        m = fmaxf(m, __shfl_xor(m, 2, 16));
        m = fmaxf(m, __shfl_xor(m, 4, 16));
        m = fmaxf(m, __shfl_xor(m, 8, 16));

        float p[9];
        float s = 0.f;
#pragma unroll
        for (int kt = 0; kt < 9; ++kt) { p[kt] = __expf(sc[kt][r] - m); s += p[kt]; }
        s += __shfl_xor(s, 1, 16);
        s += __shfl_xor(s, 2, 16);
        s += __shfl_xor(s, 4, 16);
        s += __shfl_xor(s, 8, 16);
        const float inv = 1.0f / s;

        const int mrow = hi * 8 + r;
#pragma unroll
        for (int kt = 0; kt < 9; ++kt)
            P[mrow * 160 + kt * 16 + lo16] = f2bf(p[kt] * inv);
    }
    __syncthreads();

    // out(16x64) = P(16x160) @ V^T, keys chunked by 32 (5 chunks, zero padded)
    const unsigned short* vbase = Vtg + (size_t)((b * Hc + h) * 64) * Sc;
#pragma unroll
    for (int dt = 0; dt < 4; ++dt) {
        v8f acc = {};
#pragma unroll
        for (int kc = 0; kc < 5; ++kc) {
            const unsigned short* pp = P + lo16 * 160 + kc * 32;
            FragBF fa;
            fa.h[0] = *(const v8bf*)(pp + kb);
            fa.h[1] = *(const v8bf*)(pp + kb + 16);
            const int d = dt * 16 + lo16;
            const unsigned short* vp = vbase + (size_t)d * Sc + s0 + kc * 32 + kb;
            FragBF fb;
            fb.h[0] = *(const v8bf*)(vp);
            fb.h[1] = *(const v8bf*)(vp + 16);
            acc = WMMA_BF16(fa.v, fb.v, acc);
        }
#pragma unroll
        for (int r = 0; r < 8; ++r) {
            const int row = q0 + hi * 8 + r;
            const int col = h * 64 + dt * 16 + lo16;
            Og[(size_t)(b * Sc + row) * DMc + col] = f2bf(acc[r]);
        }
    }
}

// ---------------------------------------------------------------------------
// Launcher
// ---------------------------------------------------------------------------
extern "C" void kernel_launch(void* const* d_in, const int* in_sizes, int n_in,
                              void* d_out, int out_size, void* d_ws, size_t ws_size,
                              hipStream_t stream) {
    (void)in_sizes; (void)n_in; (void)out_size; (void)ws_size;

    const float* x    = (const float*)d_in[0];   // [2,2048,768]
    const float* wqkv = (const float*)d_in[1];   // [768,2304]
    const float* wo   = (const float*)d_in[2];   // [768,768]
    float* out = (float*)d_out;                  // [2,2048,768] f32

    const int M  = Bc * Sc;       // 4096
    const int N1 = 3 * DMc;       // 2304
    const int K1 = DMc;           // 768

    char* ws = (char*)d_ws;
    size_t off = 0;
    auto take = [&](size_t bytes) -> char* {
        char* p = ws + off;
        off = (off + bytes + 4095) & ~(size_t)4095;   // pad: tolerate small tail over-reads
        return p;
    };

    unsigned short* Xb  = (unsigned short*)take((size_t)M * K1 * 2);        // x bf16
    unsigned short* Wq  = (unsigned short*)take((size_t)N1 * K1 * 2);       // wqkv^T bf16
    unsigned short* Wo  = (unsigned short*)take((size_t)DMc * DMc * 2);     // wo^T bf16
    float*          qkv = (float*)take((size_t)M * N1 * 4);                 // qkv f32
    unsigned short* Q   = (unsigned short*)take((size_t)Bc * Hc * Sc * 64 * 2);
    unsigned short* Kb  = (unsigned short*)take((size_t)Bc * Hc * Sc * 64 * 2);
    unsigned short* Vt  = (unsigned short*)take((size_t)Bc * Hc * 64 * Sc * 2);
    unsigned short* Ob  = (unsigned short*)take((size_t)M * DMc * 2);       // attn out bf16

    // 1) convert inputs to bf16 (weights transposed for WMMA B-operand)
    {
        int n = M * K1;
        k_cvt<<<(n + 255) / 256, 256, 0, stream>>>(x, Xb, n);
    }
    {
        int n = K1 * N1;
        k_transpose_cvt<<<(n + 255) / 256, 256, 0, stream>>>(wqkv, Wq, K1, N1);
    }
    {
        int n = DMc * DMc;
        k_transpose_cvt<<<(n + 255) / 256, 256, 0, stream>>>(wo, Wo, DMc, DMc);
    }

    // 2) QKV projection: qkv = Xb @ wqkv   (4096 x 2304, K=768)
    k_gemm<<<dim3(N1 / 64, M / 64), 128, 0, stream>>>(Xb, Wq, qkv, M, N1, K1);

    // 3) RoPE + scatter into Q,K [b,h,s,d] and V^T [b,h,d,s] (bf16)
    {
        int n = Bc * Sc * Hc * 32;
        k_rope<<<(n + 255) / 256, 256, 0, stream>>>(qkv, Q, Kb, Vt);
    }

    // 4) sliding-window attention: one wave per (b,h,16-query tile)
    k_attn<<<Bc * Hc * (Sc / 16), 32, 0, stream>>>(Q, Kb, Vt, Ob);

    // 5) output projection: out = Ob @ wo  (4096 x 768, K=768)
    k_gemm<<<dim3(DMc / 64, M / 64), 128, 0, stream>>>(Ob, Wo, out, M, DMc, K1);
}
